// MambaDecoder_31001073942925
// MI455X (gfx1250) — matrix operand order repaired and checked by reading, not compiled, not run on hardware
//
#include <hip/hip_runtime.h>
#include <hip/hip_bf16.h>

typedef __attribute__((ext_vector_type(16))) _Float16 v16h;
typedef __attribute__((ext_vector_type(4)))  _Float16 h4;
typedef __attribute__((ext_vector_type(8)))  float    v8f;
typedef __attribute__((ext_vector_type(4)))  float    f32x4;

// Global-address-space pointer types: force global_load/global_store codegen.
typedef __attribute__((address_space(1))) const float    GCF;
typedef __attribute__((address_space(1))) float          GF;
typedef __attribute__((address_space(1))) const f32x4    GCF4;
typedef __attribute__((address_space(1))) const _Float16 GCH;
typedef __attribute__((address_space(1))) const v16h     GCH16;
typedef __attribute__((address_space(1))) h4             GH4;

#define NEURAL_DIM 256
#define D_MODEL    512
#define D_STATE    16
#define D_CONV     4
#define D_INNER    1024
#define DT_RANK    32
#define LAYERS     4
#define NDAYS      24
#define NCLS       41          // N_CLASSES + 1
#define B_SZ       8
#define T_LEN      2000
#define STRIDE_W   4
#define KLEN       14
#define L_OUT      497         // (2000-14)/4+1
#define SMOOTH_K   20
#define M_ROWS     (B_SZ * L_OUT)   // 3976

static __device__ __forceinline__ int imin(int a, int b) { return a < b ? a : b; }

// ---------------------------------------------------------------------------
// f32 -> f16 weight conversion (vectorized 4-wide; all counts are mult. of 4)
// ---------------------------------------------------------------------------
__global__ void f32_to_f16(const float* srcp, _Float16* dstp, long n4) {
  long i = (long)blockIdx.x * blockDim.x + threadIdx.x;
  if (i >= n4) return;
  const f32x4 v = ((GCF4*)srcp)[i];
  h4 o;
  o.x = (_Float16)v.x; o.y = (_Float16)v.y;
  o.z = (_Float16)v.z; o.w = (_Float16)v.w;
  ((GH4*)dstp)[i] = o;
}

// ---------------------------------------------------------------------------
// WMMA GEMM: C = epi(A @ W^T + bias).  Wave tile: 16(M) x 64(N), 4 wmma/K-step.
// W is pre-converted f16 (loaded natively, no per-iteration conversion).
// Out-of-range A rows / W rows are CLAMPED, not guarded: WMMA element (m,n)
// depends only on A row m and B column n, and fringe (m,n) are never stored.
//   AMODE 0: A plain (M,K) f32 row-major (+batch stride)
//   AMODE 1: A gathered from x2 (B,T,256) f32 strided windows, K = 256*14
//   BMODE 0: W (N,K) f16 row-major (torch weight)  -> one v16h load per tile
//   BMODE 1: W = day_w f16 (NDAYS,256,256) math layout, selected by dayIdx[z]
//   EPI   0: none   1: +bias   2: +day_b then softsign   3: +bias then softplus
// ---------------------------------------------------------------------------
struct GemmP {
  const float*    A;
  const _Float16* W;
  const float*    bias;
  float*          C;
  const int*      dayIdx;
  long a_bstride, c_bstride;
  int M, N, K, lda, ldb, ldc;   // ldb in f16 elements
};

template <int AMODE, int BMODE, int EPI>
__global__ __launch_bounds__(128) void gemm_wmma(GemmP p) {
  const int lane  = threadIdx.x & 31;
  const int wid   = threadIdx.x >> 5;
  const int tileM = blockIdx.y * 64 + wid * 16;
  const int tileN = blockIdx.x * 64;
  const int z     = blockIdx.z;
  if (tileM >= p.M) return;

  const int hlf  = lane >> 4;
  const int mloc = lane & 15;
  const int mg   = imin(tileM + mloc, p.M - 1);   // clamped A row

  GCF* Abase = (GCF*)p.A + (long)z * p.a_bstride;
  GCH* Wbase = (GCH*)p.W;
  if constexpr (BMODE == 1)
    Wbase += (long)p.dayIdx[z] * ((long)NEURAL_DIM * NEURAL_DIM);

  // ---- A stream (plain): one base pointer, +32 floats/K-step, imm offsets ----
  GCF* ap = Abase + (long)mg * p.lda + (hlf << 4);

  // ---- B streams: 4 base pointers (one per N-subtile), imm offsets ----------
  GCH* bp[4];
#pragma unroll
  for (int t = 0; t < 4; ++t) {
    const int ng = imin(tileN + t * 16 + mloc, p.N - 1);   // clamped W row
    if constexpr (BMODE == 0) bp[t] = Wbase + (long)ng * p.ldb + (hlf << 4);
    else                      bp[t] = Wbase + (long)(hlf << 4) * NEURAL_DIM + ng;
  }

  // ---- Windowed-gather state: per-slot int offsets, branchless advance ------
  // k += 32 maps to (c += 2, j += 4) with carry when j >= 10:
  //   no carry: off += 4*256 + 2 = 1026 ; carry: off += -10*256 + 3 = -2557
  GCF* wap = nullptr;
  int woff[16], wj[16];
  if constexpr (AMODE == 1) {
    const int wb = mg / L_OUT, wl = mg - wb * L_OUT;
    wap = Abase + ((long)wb * T_LEN + (long)wl * STRIDE_W) * NEURAL_DIM;
#pragma unroll
    for (int v = 0; v < 8; ++v)
#pragma unroll
      for (int e = 0; e < 2; ++e) {
        const int s   = v * 2 + e;
        const int off = (hlf << 4) + (v << 1) + e;
        const int c   = off / KLEN;
        wj[s]   = off - c * KLEN;
        woff[s] = wj[s] * NEURAL_DIM + c;
      }
  }

  v8f acc0 = {}, acc1 = {}, acc2 = {}, acc3 = {};
  const int ksteps = p.K >> 5;              // all K are multiples of 32
  for (int kt = 0; kt < ksteps; ++kt) {
    // ---- A fragment (16x32 f16: k = hlf*16 + 2v + e) --------
    v16h af;
    if constexpr (AMODE == 0) {
#pragma unroll
      for (int g = 0; g < 2; ++g) {                    // 2 x (2 x b128)
        const f32x4 q0 = *(GCF4*)(ap + (g << 3));
        const f32x4 q1 = *(GCF4*)(ap + (g << 3) + 4);
        af[8 * g + 0] = (_Float16)q0.x; af[8 * g + 1] = (_Float16)q0.y;
        af[8 * g + 2] = (_Float16)q0.z; af[8 * g + 3] = (_Float16)q0.w;
        af[8 * g + 4] = (_Float16)q1.x; af[8 * g + 5] = (_Float16)q1.y;
        af[8 * g + 6] = (_Float16)q1.z; af[8 * g + 7] = (_Float16)q1.w;
      }
      ap += 32;
    } else {
#pragma unroll
      for (int s = 0; s < 16; ++s) {
        af[s] = (_Float16)wap[woff[s]];       // af[2v+e] with s = 2v+e
        const bool cy = wj[s] >= 10;
        woff[s] += cy ? -2557 : 1026;
        wj[s]   += cy ? -10   : 4;
      }
    }

    // ---- B fragments (32x16 f16: k = hlf*16 + 2v + e, n = lane&15) ----------
    v16h bf[4];
#pragma unroll
    for (int t = 0; t < 4; ++t) {
      if constexpr (BMODE == 0) {
        bf[t] = *(GCH16*)(bp[t]);                      // 32B: 2 x b128, no cvt
        bp[t] += 32;
      } else {
#pragma unroll
        for (int s = 0; s < 16; ++s)                   // b16 loads, imm offsets
          bf[t][s] = bp[t][s * NEURAL_DIM];
        bp[t] += 32 * NEURAL_DIM;
      }
    }

    acc0 = __builtin_amdgcn_wmma_f32_16x16x32_f16(false, af, false, bf[0], (short)0, acc0, false, false);
    acc1 = __builtin_amdgcn_wmma_f32_16x16x32_f16(false, af, false, bf[1], (short)0, acc1, false, false);
    acc2 = __builtin_amdgcn_wmma_f32_16x16x32_f16(false, af, false, bf[2], (short)0, acc2, false, false);
    acc3 = __builtin_amdgcn_wmma_f32_16x16x32_f16(false, af, false, bf[3], (short)0, acc3, false, false);
  }

  // ---- epilogue; C/D layout: n = lane&15, m = j + 8*(lane>>4) ---------------
  GCF* biasp = (GCF*)p.bias;
  if constexpr (EPI == 2) biasp += (long)p.dayIdx[z] * p.N;
  GF* Cbase = (GF*)p.C + (long)z * p.c_bstride;

  auto store_tile = [&](const v8f& a, int t) {
    const int n = tileN + t * 16 + mloc;
    if (n >= p.N) return;
    float bv = 0.f;
    if constexpr (EPI != 0) bv = biasp[n];
#pragma unroll
    for (int j = 0; j < 8; ++j) {
      const int m = tileM + j + (hlf << 3);
      if (m < p.M) {
        float val = a[j] + bv;
        if constexpr (EPI == 2)      val = val / (1.f + fabsf(val));
        else if constexpr (EPI == 3) val = (val > 20.f) ? val : log1pf(__expf(val));
        Cbase[(long)m * p.ldc + n] = val;
      }
    }
  };
  store_tile(acc0, 0); store_tile(acc1, 1); store_tile(acc2, 2); store_tile(acc3, 3);
}

// ---------------------------------------------------------------------------
// Gaussian temporal smoothing, SAME padding (pad_lo=9, pad_hi=10), sigma=2
// ---------------------------------------------------------------------------
__global__ void gauss_smooth(const float* xin, float* xout) {
  GCF* x   = (GCF*)xin;
  GF*  out = (GF*)xout;
  const long total = (long)B_SZ * T_LEN * NEURAL_DIM;
  long idx = (long)blockIdx.x * blockDim.x + threadIdx.x;
  if (idx >= total) return;
  const int c = (int)(idx % NEURAL_DIM);
  const int t = (int)((idx / NEURAL_DIM) % T_LEN);
  const int b = (int)(idx / ((long)NEURAL_DIM * T_LEN));
  float s = 0.f, accum = 0.f;
#pragma unroll
  for (int k = 0; k < SMOOTH_K; ++k) {
    const float d = ((float)k - 9.5f) * 0.5f;     // (k - mean)/sigma
    const float w = __expf(-0.5f * d * d);
    s += w;
    const int tt = t + k - 9;
    if (tt >= 0 && tt < T_LEN)
      accum += w * x[((long)b * T_LEN + tt) * NEURAL_DIM + c];
  }
  out[idx] = accum / s;
}

// ---------------------------------------------------------------------------
// Causal depthwise conv (D_CONV=4, pad (3,0)) + bias + SiLU on x-half of xz
// ---------------------------------------------------------------------------
__global__ void conv_silu(const float* xzin, const float* cwin,
                          const float* cbin, float* xoutp) {
  GCF* xz = (GCF*)xzin; GCF* cw = (GCF*)cwin; GCF* cb = (GCF*)cbin;
  GF* xout = (GF*)xoutp;
  const long total = (long)M_ROWS * D_INNER;
  long idx = (long)blockIdx.x * blockDim.x + threadIdx.x;
  if (idx >= total) return;
  const int  d = (int)(idx % D_INNER);
  const long r = idx / D_INNER;
  const int  l = (int)(r % L_OUT);
  const int  b = (int)(r / L_OUT);
  float acc = cb[d];
#pragma unroll
  for (int j = 0; j < D_CONV; ++j) {
    const int t = l - (D_CONV - 1) + j;
    if (t >= 0)
      acc += xz[((long)(b * L_OUT + t)) * (2 * D_INNER) + d] * cw[d * D_CONV + j];
  }
  xout[idx] = acc / (1.f + __expf(-acc));   // acc * sigmoid(acc)
}

// ---------------------------------------------------------------------------
// Selective-scan: per (batch, channel) lane keeps 16 states in registers;
// B/C (32 floats per step) broadcast through LDS. Then gate with silu(z).
// ---------------------------------------------------------------------------
__global__ __launch_bounds__(256) void ssm_scan(
    const float* xzin, const float* xcin, const float* xdbcin,
    const float* deltain, const float* A_login, const float* Dpin, float* yout) {
  GCF* xz = (GCF*)xzin;  GCF* xc = (GCF*)xcin;  GCF* xdbc = (GCF*)xdbcin;
  GCF* delta = (GCF*)deltain; GCF* A_log = (GCF*)A_login; GCF* Dp = (GCF*)Dpin;
  GF* y = (GF*)yout;

  const int b = blockIdx.x;
  const int d = blockIdx.y * 256 + threadIdx.x;
  __shared__ float sBC[32];
  float Avec[D_STATE];
#pragma unroll
  for (int n = 0; n < D_STATE; ++n) Avec[n] = -__expf(A_log[(long)d * D_STATE + n]);
  const float Dd = Dp[d];
  float h[D_STATE];
#pragma unroll
  for (int n = 0; n < D_STATE; ++n) h[n] = 0.f;

  for (int l = 0; l < L_OUT; ++l) {
    const long row = (long)b * L_OUT + l;
    if (threadIdx.x < 32) sBC[threadIdx.x] = xdbc[row * 64 + DT_RANK + threadIdx.x];
    __syncthreads();
    const float dlt = delta[row * D_INNER + d];
    const float xv  = xc[row * D_INNER + d];
    const float dx  = dlt * xv;
    float yv = 0.f;
#pragma unroll
    for (int n = 0; n < D_STATE; ++n) {
      h[n] = __expf(dlt * Avec[n]) * h[n] + dx * sBC[n];
      yv  += h[n] * sBC[16 + n];
    }
    const float zv = xz[row * (2 * D_INNER) + D_INNER + d];
    y[row * D_INNER + d] = (yv + xv * Dd) * (zv / (1.f + __expf(-zv)));
    __syncthreads();
  }
}

// ---------------------------------------------------------------------------
static dim3 gemm_grid(const GemmP& p, int batches) {
  return dim3((p.N + 63) / 64, (p.M + 63) / 64, batches);
}
static void cvt_w(hipStream_t s, const float* src, _Float16* dst, long n) {
  const long n4 = n / 4;
  f32_to_f16<<<(unsigned)((n4 + 255) / 256), 256, 0, s>>>(src, dst, n4);
}

extern "C" void kernel_launch(void* const* d_in, const int* in_sizes, int n_in,
                              void* d_out, int out_size, void* d_ws, size_t ws_size,
                              hipStream_t stream) {
  (void)in_sizes; (void)n_in; (void)out_size; (void)ws_size;
  const float* neural     = (const float*)d_in[0];
  const int*   dayIdx     = (const int*)  d_in[1];
  const float* day_w      = (const float*)d_in[2];
  const float* day_b      = (const float*)d_in[3];
  const float* lin_in_w   = (const float*)d_in[4];
  const float* lin_in_b   = (const float*)d_in[5];
  const float* in_proj_w  = (const float*)d_in[6];
  const float* conv_w     = (const float*)d_in[7];
  const float* conv_b     = (const float*)d_in[8];
  const float* x_proj_w   = (const float*)d_in[9];
  const float* dt_w       = (const float*)d_in[10];
  const float* dt_b       = (const float*)d_in[11];
  const float* A_log      = (const float*)d_in[12];
  const float* D_param    = (const float*)d_in[13];
  const float* out_proj_w = (const float*)d_in[14];
  const float* fc_w       = (const float*)d_in[15];
  const float* fc_b       = (const float*)d_in[16];
  float* ws = (float*)d_ws;

  // workspace (floats); xz reuses the region of smoothed+x2 once they're dead
  const long SM_OFF = 0;
  const long X2_OFF = (long)B_SZ * T_LEN * NEURAL_DIM;        // 4,096,000
  const long XZ_OFF = 0;                                      // reuses [0, 8.19M)
  const long H_OFF  = 2L * B_SZ * T_LEN * NEURAL_DIM;         // 8,192,000
  const long XC_OFF = H_OFF  + (long)M_ROWS * D_MODEL;
  const long XD_OFF = XC_OFF + (long)M_ROWS * D_INNER;
  const long DL_OFF = XD_OFF + (long)M_ROWS * 64;
  const long Y_OFF  = DL_OFF + (long)M_ROWS * D_INNER;
  const long WH_OFF = Y_OFF  + (long)M_ROWS * D_INNER;        // f16 weights

  // f16 weight sub-offsets (in halves; all multiples of 16 -> 32B aligned)
  _Float16* wh = (_Float16*)(ws + WH_OFF);
  const long DAYW_H = 0;
  const long LINW_H = DAYW_H + (long)NDAYS * NEURAL_DIM * NEURAL_DIM;  // 1,572,864
  const long IPW_H  = LINW_H + (long)D_MODEL * NEURAL_DIM * KLEN;      // +1,835,008
  const long XPW_H  = IPW_H  + (long)LAYERS * 2 * D_INNER * D_MODEL;   // +4,194,304
  const long DTW_H  = XPW_H  + (long)LAYERS * 64 * D_INNER;            // +262,144
  const long OPW_H  = DTW_H  + (long)LAYERS * D_INNER * DT_RANK;       // +131,072
  const long FCW_H  = OPW_H  + (long)LAYERS * D_MODEL * D_INNER;       // +2,097,152

  // 0) convert all GEMM weights to f16 once per call
  cvt_w(stream, day_w,      wh + DAYW_H, (long)NDAYS * NEURAL_DIM * NEURAL_DIM);
  cvt_w(stream, lin_in_w,   wh + LINW_H, (long)D_MODEL * NEURAL_DIM * KLEN);
  cvt_w(stream, in_proj_w,  wh + IPW_H,  (long)LAYERS * 2 * D_INNER * D_MODEL);
  cvt_w(stream, x_proj_w,   wh + XPW_H,  (long)LAYERS * 64 * D_INNER);
  cvt_w(stream, dt_w,       wh + DTW_H,  (long)LAYERS * D_INNER * DT_RANK);
  cvt_w(stream, out_proj_w, wh + OPW_H,  (long)LAYERS * D_MODEL * D_INNER);
  cvt_w(stream, fc_w,       wh + FCW_H,  (long)NCLS * D_MODEL);

  // 1) Gaussian smoothing
  {
    const long tot = (long)B_SZ * T_LEN * NEURAL_DIM;
    gauss_smooth<<<(unsigned)((tot + 255) / 256), 256, 0, stream>>>(neural, ws + SM_OFF);
  }
  // 2) Day transform: x2 = softsign(smoothed @ day_w[dayIdx[b]] + day_b[dayIdx[b]])
  {
    GemmP p{};
    p.A = ws + SM_OFF; p.W = wh + DAYW_H; p.bias = day_b; p.C = ws + X2_OFF;
    p.dayIdx = dayIdx;
    p.a_bstride = (long)T_LEN * NEURAL_DIM; p.c_bstride = (long)T_LEN * NEURAL_DIM;
    p.M = T_LEN; p.N = NEURAL_DIM; p.K = NEURAL_DIM;
    p.lda = NEURAL_DIM; p.ldb = 0; p.ldc = NEURAL_DIM;
    gemm_wmma<0, 1, 2><<<gemm_grid(p, B_SZ), 128, 0, stream>>>(p);
  }
  // 3) Windowed lin_in: h = strided(x2) @ lin_in_w.T + b   (gather fused into A)
  {
    GemmP p{};
    p.A = ws + X2_OFF; p.W = wh + LINW_H; p.bias = lin_in_b; p.C = ws + H_OFF;
    p.M = M_ROWS; p.N = D_MODEL; p.K = NEURAL_DIM * KLEN;
    p.lda = 0; p.ldb = NEURAL_DIM * KLEN; p.ldc = D_MODEL;
    gemm_wmma<1, 0, 1><<<gemm_grid(p, 1), 128, 0, stream>>>(p);
  }
  // 4) Mamba blocks
  for (int i = 0; i < LAYERS; ++i) {
    const _Float16* ipw = wh + IPW_H + (long)i * 2 * D_INNER * D_MODEL;
    const _Float16* xpw = wh + XPW_H + (long)i * 64 * D_INNER;
    const _Float16* dtw = wh + DTW_H + (long)i * D_INNER * DT_RANK;
    const _Float16* opw = wh + OPW_H + (long)i * D_MODEL * D_INNER;
    const float* cw  = conv_w  + (long)i * D_INNER * D_CONV;
    const float* cb  = conv_b  + (long)i * D_INNER;
    const float* dtb = dt_b    + (long)i * D_INNER;
    const float* al  = A_log   + (long)i * D_INNER * D_STATE;
    const float* dp  = D_param + (long)i * D_INNER;

    { // in_proj: xz = h @ ipw.T
      GemmP p{};
      p.A = ws + H_OFF; p.W = ipw; p.C = ws + XZ_OFF;
      p.M = M_ROWS; p.N = 2 * D_INNER; p.K = D_MODEL;
      p.lda = D_MODEL; p.ldb = D_MODEL; p.ldc = 2 * D_INNER;
      gemm_wmma<0, 0, 0><<<gemm_grid(p, 1), 128, 0, stream>>>(p);
    }
    { // causal depthwise conv + SiLU
      const long tot = (long)M_ROWS * D_INNER;
      conv_silu<<<(unsigned)((tot + 255) / 256), 256, 0, stream>>>(ws + XZ_OFF, cw, cb, ws + XC_OFF);
    }
    { // x_proj: xdbc = x @ xpw.T  (dt | B | C)
      GemmP p{};
      p.A = ws + XC_OFF; p.W = xpw; p.C = ws + XD_OFF;
      p.M = M_ROWS; p.N = 64; p.K = D_INNER;
      p.lda = D_INNER; p.ldb = D_INNER; p.ldc = 64;
      gemm_wmma<0, 0, 0><<<gemm_grid(p, 1), 128, 0, stream>>>(p);
    }
    { // delta = softplus(dt @ dtw.T + dtb)  (A = first 32 cols of xdbc, lda=64)
      GemmP p{};
      p.A = ws + XD_OFF; p.W = dtw; p.bias = dtb; p.C = ws + DL_OFF;
      p.M = M_ROWS; p.N = D_INNER; p.K = DT_RANK;
      p.lda = 64; p.ldb = DT_RANK; p.ldc = D_INNER;
      gemm_wmma<0, 0, 3><<<gemm_grid(p, 1), 128, 0, stream>>>(p);
    }
    { // selective scan + D skip + silu(z) gate
      dim3 g(B_SZ, D_INNER / 256);
      ssm_scan<<<g, 256, 0, stream>>>(ws + XZ_OFF, ws + XC_OFF, ws + XD_OFF,
                                      ws + DL_OFF, al, dp, ws + Y_OFF);
    }
    { // out_proj: h = y @ opw.T
      GemmP p{};
      p.A = ws + Y_OFF; p.W = opw; p.C = ws + H_OFF;
      p.M = M_ROWS; p.N = D_MODEL; p.K = D_INNER;
      p.lda = D_INNER; p.ldb = D_INNER; p.ldc = D_MODEL;
      gemm_wmma<0, 0, 0><<<gemm_grid(p, 1), 128, 0, stream>>>(p);
    }
  }
  // 5) classifier: out = h @ fc_w.T + fc_b
  {
    GemmP p{};
    p.A = ws + H_OFF; p.W = wh + FCW_H; p.bias = fc_b; p.C = (float*)d_out;
    p.M = M_ROWS; p.N = NCLS; p.K = D_MODEL;
    p.lda = D_MODEL; p.ldb = D_MODEL; p.ldc = NCLS;
    gemm_wmma<0, 0, 1><<<gemm_grid(p, 1), 128, 0, stream>>>(p);
  }
}
